// Masked_multi_head_attention_57921928954353
// MI455X (gfx1250) — matrix operand order repaired and checked by reading, not compiled, not run on hardware
//
#include <hip/hip_runtime.h>

#define H_  16
#define B_  4
#define T_  2048
#define E_  1024
#define D_  64
#define M_  (B_ * T_)     // 8192 tokens

typedef __attribute__((ext_vector_type(16))) __bf16   v16bf;
typedef __attribute__((ext_vector_type(8)))  float    v8f;
typedef __attribute__((ext_vector_type(4)))  unsigned int v4u;

union Frag { v4u u[2]; v16bf v; };   // 32 bytes: one bf16 WMMA A/B operand

__device__ __forceinline__ unsigned short f2bf(float f) {
    unsigned int u = __builtin_bit_cast(unsigned int, f);
    u += 0x7FFFu + ((u >> 16) & 1u);          // round-to-nearest-even
    return (unsigned short)(u >> 16);
}

__device__ __forceinline__ v8f wmma_bf16(const Frag& a, const Frag& b, v8f c) {
    return __builtin_amdgcn_wmma_f32_16x16x32_bf16(
        false, a.v, false, b.v, (short)0, c, false, false);
}

// Async DMA: 16 bytes per lane, global -> LDS, tracked by ASYNCcnt.
__device__ __forceinline__ void async_ld16(unsigned int lds_off,
                                           const unsigned short* g) {
    unsigned long long ga = (unsigned long long)(uintptr_t)g;
    asm volatile("global_load_async_to_lds_b128 %0, %1, off"
                 :: "v"(lds_off), "v"(ga) : "memory");
}
__device__ __forceinline__ unsigned int lds_off(const void* p) {
    return (unsigned int)(uintptr_t)p;   // low 32 bits = LDS byte offset
}

// ---------------------------------------------------------------- fp32 -> bf16
__global__ void k_cvt(const float* __restrict__ s, unsigned short* __restrict__ d, int n) {
    int i = blockIdx.x * blockDim.x + threadIdx.x;
    int stride = gridDim.x * blockDim.x;
    for (; i < n; i += stride) d[i] = f2bf(s[i]);
}

// ------------------------------------------------- QKV projection: Y = X @ W^T
// X:[M,E] bf16 row-major, W:[E,E] bf16 row-major (W[n,k]) -> both K-contiguous.
// MODE 0: write [B,H,T,D] (Q,K).  MODE 1: write transposed [B,H,D,T] (V).
// Block = 256 thr = 8 waves (2x4), wave tile 64x32, block tile 128x128.
template <int MODE>
__global__ void k_gemm_qkv(const unsigned short* __restrict__ X,
                           const unsigned short* __restrict__ W,
                           unsigned short* __restrict__ Out) {
    const int lane = threadIdx.x & 31;
    const int w    = threadIdx.x >> 5;
    const int l16  = lane & 15;
    const int hi   = lane >> 4;
    const int rowbase = blockIdx.x * 128 + (w & 1) * 64;
    const int colbase = blockIdx.y * 128 + (w >> 1) * 32;

    const v8f zero = {0.f,0.f,0.f,0.f,0.f,0.f,0.f,0.f};
    v8f acc[4][2];
#pragma unroll
    for (int i = 0; i < 4; ++i)
#pragma unroll
        for (int j = 0; j < 2; ++j) acc[i][j] = zero;

    for (int k0 = 0; k0 < E_; k0 += 32) {
        if (k0 + 32 < E_) {                       // hint next K-tile into caches
            __builtin_prefetch(X + (size_t)(rowbase + l16) * E_ + k0 + 32, 0, 3);
            __builtin_prefetch(W + (size_t)(colbase + l16) * E_ + k0 + 32, 0, 3);
        }
        Frag a[4], b[2];
#pragma unroll
        for (int i = 0; i < 4; ++i) {           // A 16x32: lane row, K = {0|8}+0..7 and +16
            int m = rowbase + i * 16 + l16;
            const v4u* p = (const v4u*)(X + (size_t)m * E_ + k0 + hi * 8);
            a[i].u[0] = p[0];
            a[i].u[1] = p[2];
        }
#pragma unroll
        for (int j = 0; j < 2; ++j) {           // B 32x16: lane col, K = {0|8}+0..7 and +16
            int n = colbase + j * 16 + l16;
            const v4u* p = (const v4u*)(W + (size_t)n * E_ + k0 + hi * 8);
            b[j].u[0] = p[0];
            b[j].u[1] = p[2];
        }
#pragma unroll
        for (int i = 0; i < 4; ++i)
#pragma unroll
            for (int j = 0; j < 2; ++j) acc[i][j] = wmma_bf16(a[i], b[j], acc[i][j]);
    }

#pragma unroll
    for (int i = 0; i < 4; ++i)
#pragma unroll
        for (int j = 0; j < 2; ++j)
#pragma unroll
            for (int v = 0; v < 8; ++v) {
                int m  = rowbase + i * 16 + v + hi * 8;     // C layout row
                int e  = colbase + j * 16 + l16;            // C layout col
                int bb = m >> 11, t = m & (T_ - 1);
                int h  = e >> 6,  dd = e & (D_ - 1);
                size_t addr = (MODE == 0)
                    ? ((size_t)(bb * H_ + h) * T_ + t) * D_ + dd      // [B,H,T,D]
                    : ((size_t)(bb * H_ + h) * D_ + dd) * T_ + t;     // [B,H,D,T]
                Out[addr] = f2bf(acc[i][j][v]);
            }
}

// ----------------------------------------------------- causal flash attention
// grid = (T/128, B*H); block = 256 thr = 8 waves; wave owns 16 query rows.
// K/V tiles for the whole block are staged into LDS with async DMA,
// double-buffered so the next tile's copy overlaps the current tile's WMMAs.
__global__ void k_attn(const unsigned short* __restrict__ Q,
                       const unsigned short* __restrict__ K,
                       const unsigned short* __restrict__ Vt,  // [B,H,D,T]
                       unsigned short* __restrict__ C) {       // [B,T,E] bf16
    __shared__ __attribute__((aligned(16))) unsigned short Kb_s[2][32 * 64]; // [key][d]
    __shared__ __attribute__((aligned(16))) unsigned short Vb_s[2][64 * 32]; // [d][key]
    __shared__ __attribute__((aligned(16))) unsigned short pbuf[8 * 16 * 32];
    const int tid  = threadIdx.x;
    const int lane = tid & 31;
    const int w    = tid >> 5;
    const int l16  = lane & 15;
    const int hi   = lane >> 4;
    const int bh   = blockIdx.y;
    const int qbase = blockIdx.x * 128 + w * 16;

    const unsigned short* Qh = Q  + (size_t)bh * T_ * D_;
    const unsigned short* Kh = K  + (size_t)bh * T_ * D_;
    const unsigned short* Vh = Vt + (size_t)bh * D_ * T_;
    unsigned short* pw = pbuf + w * (16 * 32);

    // Q fragments (A layout), D split into two 32-wide K-chunks
    Frag qf[2];
    {
        int qrow = qbase + l16;
#pragma unroll
        for (int dc = 0; dc < 2; ++dc) {
            const v4u* p = (const v4u*)(Qh + (size_t)qrow * D_ + dc * 32 + hi * 8);
            qf[dc].u[0] = p[0];
            qf[dc].u[1] = p[2];
        }
    }

    const v8f zero = {0.f,0.f,0.f,0.f,0.f,0.f,0.f,0.f};
    v8f acc[4];                       // ctx 16x64 as four 16x16 C tiles
#pragma unroll
    for (int dc = 0; dc < 4; ++dc) acc[dc] = zero;
    float mst[8], lst[8];
#pragma unroll
    for (int v = 0; v < 8; ++v) { mst[v] = -3.0e38f; lst[v] = 0.f; }

    // async copy of one 32-key chunk: K tile 32x64 (4KB) + V^T tile 64x32 (4KB)
    auto issue = [&](int buf, int kb) {
        const int kbb = kb * 32;
        {   // K: each thread copies 16B; 256 thr x 16B = 4KB tile
            int r = tid >> 3, c = (tid & 7) * 8;
            async_ld16(lds_off(&Kb_s[buf][r * 64 + c]),
                       Kh + (size_t)(kbb + r) * D_ + c);
        }
        {   // V^T
            int d = tid >> 2, c = (tid & 3) * 8;
            async_ld16(lds_off(&Vb_s[buf][d * 32 + c]),
                       Vh + (size_t)d * T_ + kbb + c);
        }
    };

    const int nmax = blockIdx.x * 4 + 4;           // chunks needed by last wave
    const int nch  = (qbase + 16 + 31) >> 5;       // chunks this wave needs

    issue(0, 0);
    for (int kb = 0; kb < nmax; ++kb) {
        const int buf = kb & 1;
        const bool pre = (kb + 1 < nmax);
        if (pre) {
            issue(buf ^ 1, kb + 1);                // prefetch next tile
            asm volatile("s_wait_asynccnt 0x2" ::: "memory");  // current tile done
        } else {
            asm volatile("s_wait_asynccnt 0x0" ::: "memory");
        }
        __syncthreads();                           // all waves' copies visible

        if (kb < nch) {
            const int kbb = kb * 32;
            const unsigned short* KT = Kb_s[buf];
            const unsigned short* VT = Vb_s[buf];

            // ---- S = (Q K^T) / 8 over a 16x32 key chunk: 4 WMMA
            Frag kf[2][2];                         // [d-chunk][n-half]
#pragma unroll
            for (int nk = 0; nk < 2; ++nk) {
                int kk = nk * 16 + l16;
#pragma unroll
                for (int dc = 0; dc < 2; ++dc) {
                    const v4u* p = (const v4u*)(KT + kk * 64 + dc * 32 + hi * 8);
                    kf[dc][nk].u[0] = p[0];
                    kf[dc][nk].u[1] = p[2];
                }
            }
            v8f s[2];
#pragma unroll
            for (int nk = 0; nk < 2; ++nk) {
                v8f z = zero;
                z = wmma_bf16(qf[0], kf[0][nk], z);
                z = wmma_bf16(qf[1], kf[1][nk], z);
                s[nk] = z;
            }

            // ---- scale + causal mask
#pragma unroll
            for (int nk = 0; nk < 2; ++nk) {
                int kk = kbb + nk * 16 + l16;
#pragma unroll
                for (int v = 0; v < 8; ++v) {
                    int q = qbase + v + hi * 8;
                    float val = s[nk][v] * 0.125f;           // 1/sqrt(64)
                    s[nk][v] = (kk > q) ? -1.0e9f : val;
                }
            }

            // ---- online softmax: row max / rescale / exp / row sum
            float alpha[8];
#pragma unroll
            for (int v = 0; v < 8; ++v) {
                float t = fmaxf(s[0][v], s[1][v]);
#pragma unroll
                for (int off = 1; off < 16; off <<= 1)
                    t = fmaxf(t, __shfl_xor(t, off, 32));    // 16-lane half reduce
                float nm = fmaxf(mst[v], t);
                alpha[v] = __expf(mst[v] - nm);
                mst[v] = nm;
            }
#pragma unroll
            for (int nk = 0; nk < 2; ++nk)
#pragma unroll
                for (int v = 0; v < 8; ++v)
                    s[nk][v] = __expf(s[nk][v] - mst[v]);
#pragma unroll
            for (int v = 0; v < 8; ++v) {
                float t = s[0][v] + s[1][v];
#pragma unroll
                for (int off = 1; off < 16; off <<= 1)
                    t += __shfl_xor(t, off, 32);
                lst[v] = lst[v] * alpha[v] + t;
            }
#pragma unroll
            for (int dc = 0; dc < 4; ++dc)
#pragma unroll
                for (int v = 0; v < 8; ++v) acc[dc][v] *= alpha[v];

            // ---- C-layout P -> LDS -> A-layout reload (same-wave LDS in-order)
#pragma unroll
            for (int nk = 0; nk < 2; ++nk)
#pragma unroll
                for (int v = 0; v < 8; ++v)
                    pw[(v + hi * 8) * 32 + nk * 16 + l16] = f2bf(s[nk][v]);
            Frag pf;
            {
                const v4u* p = (const v4u*)(pw + l16 * 32 + hi * 8);
                pf.u[0] = p[0];
                pf.u[1] = p[2];
            }

            // ---- ctx += P @ V : 4 WMMA (V^T LDS tile, contiguous b128 loads)
#pragma unroll
            for (int dc = 0; dc < 4; ++dc) {
                Frag vf;
                const v4u* p = (const v4u*)(VT + (dc * 16 + l16) * 32 + hi * 8);
                vf.u[0] = p[0];
                vf.u[1] = p[2];
                acc[dc] = wmma_bf16(pf, vf, acc[dc]);
            }
        }
        __syncthreads();    // everyone done reading buf before it is re-filled
    }

    // ---- normalize and write ctx to [B,T,E] bf16
    const int bb = bh >> 4, hh = bh & 15;
#pragma unroll
    for (int dc = 0; dc < 4; ++dc)
#pragma unroll
        for (int v = 0; v < 8; ++v) {
            int q = qbase + v + hi * 8;
            int d = dc * 16 + l16;
            C[((size_t)(bb * T_ + q)) * E_ + hh * D_ + d] = f2bf(acc[dc][v] / lst[v]);
        }
}

// ------------------------------------------- output projection + bias (fp32 out)
__global__ void k_gemm_out(const unsigned short* __restrict__ X,
                           const unsigned short* __restrict__ W,
                           const float* __restrict__ bias,
                           float* __restrict__ Out) {
    const int lane = threadIdx.x & 31;
    const int w    = threadIdx.x >> 5;
    const int l16  = lane & 15;
    const int hi   = lane >> 4;
    const int rowbase = blockIdx.x * 128 + (w & 1) * 64;
    const int colbase = blockIdx.y * 128 + (w >> 1) * 32;

    const v8f zero = {0.f,0.f,0.f,0.f,0.f,0.f,0.f,0.f};
    v8f acc[4][2];
#pragma unroll
    for (int i = 0; i < 4; ++i)
#pragma unroll
        for (int j = 0; j < 2; ++j) acc[i][j] = zero;

    for (int k0 = 0; k0 < E_; k0 += 32) {
        if (k0 + 32 < E_) {
            __builtin_prefetch(X + (size_t)(rowbase + l16) * E_ + k0 + 32, 0, 3);
            __builtin_prefetch(W + (size_t)(colbase + l16) * E_ + k0 + 32, 0, 3);
        }
        Frag a[4], b[2];
#pragma unroll
        for (int i = 0; i < 4; ++i) {
            int m = rowbase + i * 16 + l16;
            const v4u* p = (const v4u*)(X + (size_t)m * E_ + k0 + hi * 8);
            a[i].u[0] = p[0];
            a[i].u[1] = p[2];
        }
#pragma unroll
        for (int j = 0; j < 2; ++j) {
            int n = colbase + j * 16 + l16;
            const v4u* p = (const v4u*)(W + (size_t)n * E_ + k0 + hi * 8);
            b[j].u[0] = p[0];
            b[j].u[1] = p[2];
        }
#pragma unroll
        for (int i = 0; i < 4; ++i)
#pragma unroll
            for (int j = 0; j < 2; ++j) acc[i][j] = wmma_bf16(a[i], b[j], acc[i][j]);
    }

    float bv[2];
#pragma unroll
    for (int j = 0; j < 2; ++j) bv[j] = bias[colbase + j * 16 + l16];

#pragma unroll
    for (int i = 0; i < 4; ++i)
#pragma unroll
        for (int j = 0; j < 2; ++j)
#pragma unroll
            for (int v = 0; v < 8; ++v) {
                int m = rowbase + i * 16 + v + hi * 8;
                int e = colbase + j * 16 + l16;
                Out[(size_t)m * E_ + e] = acc[i][j][v] + bv[j];
            }
}

// ----------------------------------------------------------------------------
extern "C" void kernel_launch(void* const* d_in, const int* in_sizes, int n_in,
                              void* d_out, int out_size, void* d_ws, size_t ws_size,
                              hipStream_t stream) {
    const float* x  = (const float*)d_in[0];
    const float* Wq = (const float*)d_in[1];
    const float* Wk = (const float*)d_in[2];
    const float* Wv = (const float*)d_in[3];
    const float* Wo = (const float*)d_in[4];
    const float* bo = (const float*)d_in[5];
    float* out = (float*)d_out;

    // workspace carve-out (bf16 everywhere): 88 MiB total
    char* p = (char*)d_ws;
    unsigned short* xb  = (unsigned short*)p; p += (size_t)M_ * E_ * 2;
    unsigned short* wqb = (unsigned short*)p; p += (size_t)E_ * E_ * 2;
    unsigned short* wkb = (unsigned short*)p; p += (size_t)E_ * E_ * 2;
    unsigned short* wvb = (unsigned short*)p; p += (size_t)E_ * E_ * 2;
    unsigned short* wob = (unsigned short*)p; p += (size_t)E_ * E_ * 2;
    unsigned short* qb  = (unsigned short*)p; p += (size_t)M_ * E_ * 2;  // [B,H,T,D]
    unsigned short* kb  = (unsigned short*)p; p += (size_t)M_ * E_ * 2;  // [B,H,T,D]
    unsigned short* vt  = (unsigned short*)p; p += (size_t)M_ * E_ * 2;  // [B,H,D,T]
    unsigned short* cb  = (unsigned short*)p; p += (size_t)M_ * E_ * 2;  // [B,T,E]

    k_cvt<<<2048, 256, 0, stream>>>(x,  xb,  M_ * E_);
    k_cvt<<<1024, 256, 0, stream>>>(Wq, wqb, E_ * E_);
    k_cvt<<<1024, 256, 0, stream>>>(Wk, wkb, E_ * E_);
    k_cvt<<<1024, 256, 0, stream>>>(Wv, wvb, E_ * E_);
    k_cvt<<<1024, 256, 0, stream>>>(Wo, wob, E_ * E_);

    dim3 gg(M_ / 128, E_ / 128);
    k_gemm_qkv<0><<<gg, 256, 0, stream>>>(xb, wqb, qb);
    k_gemm_qkv<0><<<gg, 256, 0, stream>>>(xb, wkb, kb);
    k_gemm_qkv<1><<<gg, 256, 0, stream>>>(xb, wvb, vt);

    dim3 ga(T_ / 128, B_ * H_);
    k_attn<<<ga, 256, 0, stream>>>(qb, kb, vt, cb);

    k_gemm_out<<<gg, 256, 0, stream>>>(cb, wob, bo, out);
}
